// WeatherPrediction_65085934403995
// MI455X (gfx1250) — matrix operand order repaired and checked by reading, not compiled, not run on hardware
//
#include <hip/hip_runtime.h>

// ---------------- problem constants ----------------
#define N_LAT     121
#define N_LON     240
#define N_SPATIAL 29040
#define N_SPHERE  2883
#define N_FEAT    78
#define LATENT    256
#define N_EDGES   17298
#define EPS_LN    1e-5f
// sin^2(deg2rad(3)/2) = sin^2(1.5 deg): dist<=3deg  <=>  a <= this
#define A_THRESH  6.8523263e-04f
#define PI_F      3.14159265358979f
#define R2D_F     57.29577951308232f

// ---------------- WMMA types ----------------
typedef __attribute__((ext_vector_type(16))) __bf16        v16bf;
typedef __attribute__((ext_vector_type(8)))  float         v8f;
typedef __attribute__((ext_vector_type(4)))  unsigned int  v4u;

union Frag { v16bf v; v4u q[2]; };   // 32 bytes; trivially copyable (union-safe)

__device__ __forceinline__ unsigned short f2bf(float x) {
  unsigned int u = __float_as_uint(x);
  u += 0x7FFFu + ((u >> 16) & 1u);           // round-to-nearest-even
  return (unsigned short)(u >> 16);
}

// ---------------- geometry kernels ----------------
__global__ void sphere_geom_kernel(float* __restrict__ sg, float* __restrict__ pos) {
  int i = blockIdx.x * blockDim.x + threadIdx.x;
  if (i >= N_SPHERE) return;
  float fi = (float)i;
  const float golden = 1.6180339887498949f;
  float theta = 2.f * PI_F * fi / golden;
  float phi   = acosf(1.f - 2.f * (fi + 0.5f) / (float)N_SPHERE);
  float sp = sinf(phi), cp = cosf(phi);
  float x = cosf(theta) * sp, y = sinf(theta) * sp, z = cp;
  pos[i*3+0] = x; pos[i*3+1] = y; pos[i*3+2] = z;
  float latr = asinf(fminf(fmaxf(z, -1.f), 1.f));
  float lonr = atan2f(y, x);
  sg[i*6+0] = latr * R2D_F;
  sg[i*6+1] = lonr * R2D_F;
  sg[i*6+2] = sinf(latr);
  sg[i*6+3] = cosf(latr);
  sg[i*6+4] = sinf(lonr);
  sg[i*6+5] = cosf(lonr);
}

__global__ void spatial_geom_kernel(float* __restrict__ pg) {
  int p = blockIdx.x * blockDim.x + threadIdx.x;
  if (p >= N_SPATIAL) return;
  int i = p / N_LON, j = p % N_LON;
  float lat = -90.f + (float)i * (180.f / 120.f);
  float lon = -180.f + (float)j * (360.f / 239.f);
  float latr = lat * (PI_F / 180.f), lonr = lon * (PI_F / 180.f);
  pg[p*6+0] = lat;
  pg[p*6+1] = lon;
  pg[p*6+2] = sinf(latr);
  pg[p*6+3] = cosf(latr);
  pg[p*6+4] = sinf(lonr);
  pg[p*6+5] = cosf(lonr);
}

// a = sin^2(dlat/2) + cos(sla)cos(pla) sin^2(dlon/2), via (1-cos d)/2 identities
__device__ __forceinline__ float pair_a(const float* sgrow, const float* pgrow) {
  float cdlat = sgrow[3]*pgrow[3] + sgrow[2]*pgrow[2];
  float cdlon = sgrow[5]*pgrow[5] + sgrow[4]*pgrow[4];
  return 0.5f*(1.f - cdlat) + sgrow[3]*pgrow[3]*0.5f*(1.f - cdlon);
}

__global__ void count_edges_kernel(const float* __restrict__ sg, const float* __restrict__ pg,
                                   int* __restrict__ counts) {
  int s = blockIdx.x * blockDim.x + threadIdx.x;
  if (s >= N_SPHERE) return;
  float srow[6];
  for (int t = 0; t < 6; ++t) srow[t] = sg[s*6+t];
  int cnt = 0;
  for (int p = 0; p < N_SPATIAL; ++p)
    if (pair_a(srow, &pg[p*6]) <= A_THRESH) cnt++;
  counts[s] = cnt;
}

__global__ void scan_kernel(const int* __restrict__ counts, int* __restrict__ offsets) {
  if (blockIdx.x == 0 && threadIdx.x == 0) {
    int acc = 0;
    for (int s = 0; s < N_SPHERE; ++s) { offsets[s] = acc; acc += counts[s]; }
    offsets[N_SPHERE] = acc;
  }
}

__global__ void init_idx_kernel(int* __restrict__ sphIdx, int* __restrict__ spaIdx) {
  int e = blockIdx.x * blockDim.x + threadIdx.x;
  if (e < N_EDGES) { sphIdx[e] = 0; spaIdx[e] = 0; }   // jnp.where fill_value=0
}

__global__ void write_edges_kernel(const float* __restrict__ sg, const float* __restrict__ pg,
                                   const int* __restrict__ offsets,
                                   int* __restrict__ sphIdx, int* __restrict__ spaIdx) {
  int s = blockIdx.x * blockDim.x + threadIdx.x;
  if (s >= N_SPHERE) return;
  float srow[6];
  for (int t = 0; t < 6; ++t) srow[t] = sg[s*6+t];
  int w = offsets[s];
  if (w >= N_EDGES) return;
  for (int p = 0; p < N_SPATIAL; ++p) {
    if (pair_a(srow, &pg[p*6]) <= A_THRESH) {
      if (w < N_EDGES) { sphIdx[w] = s; spaIdx[w] = p; }
      if (++w >= N_EDGES) return;
    }
  }
}

__global__ void edge_geo_kernel(const float* __restrict__ sg, const float* __restrict__ pg,
                                const int* __restrict__ sphIdx, const int* __restrict__ spaIdx,
                                float* __restrict__ dlat, float* __restrict__ dlon) {
  int e = blockIdx.x * blockDim.x + threadIdx.x;
  if (e >= N_EDGES) return;
  int s = sphIdx[e], p = spaIdx[e];
  dlat[e] = sg[s*6+0] - pg[p*6+0];
  dlon[e] = sg[s*6+1] - pg[p*6+1];
}

__global__ void neighbors_kernel(const float* __restrict__ pos, int* __restrict__ nbr) {
  int s = blockIdx.x * blockDim.x + threadIdx.x;
  if (s >= N_SPHERE) return;
  float px = pos[s*3], py = pos[s*3+1], pz = pos[s*3+2];
  float bd[7]; int bi[7];
  for (int k = 0; k < 7; ++k) { bd[k] = 1e30f; bi[k] = 0; }
  for (int j = 0; j < N_SPHERE; ++j) {
    float dot = px*pos[j*3] + py*pos[j*3+1] + pz*pos[j*3+2];
    dot = fminf(fmaxf(dot, -1.f), 1.f);
    float d = acosf(dot);
    if (d < bd[6]) {                  // strict '<' keeps earlier index on ties (top_k stability)
      int k = 6;
      while (k > 0 && d < bd[k-1]) { bd[k] = bd[k-1]; bi[k] = bi[k-1]; --k; }
      bd[k] = d; bi[k] = j;
    }
  }
  for (int k = 1; k < 7; ++k) nbr[s*6 + (k-1)] = bi[k];  // drop self (d=0)
}

// ------------- fused gather-GEMM: C[M,256] = concat(A0|A1[i1]|A2[i2]) @ W + ... -------------
// Row gr of the implicit A matrix is: A0[gr, 0:w0] ++ A1[i1[gr], 0:256] ++ A2[i2[gr], 0:256]
// (i1/i2 == nullptr means identity). bf16 inputs, f32 accumulate, v_wmma_f32_16x16x32_bf16.
// Block: 256 threads (8 waves). Tile: (MT*16) rows x 256 cols; wave w owns cols [32w,32w+32)
// as two 16x16 WMMA tiles per 16-row sub-tile -> 2*MT WMMAs per K-step.
template <int MT>
__global__ __launch_bounds__(256)
void gemm_fused(const float* __restrict__ A0, int w0,
                const float* __restrict__ A1, const int* __restrict__ i1,
                const float* __restrict__ A2, const int* __restrict__ i2,
                const float* __restrict__ W, const float* __restrict__ bias,
                const float* __restrict__ addin, float* __restrict__ C,
                int M, int K, int relu) {
  __shared__ __align__(16) unsigned short lA[MT * 16 * 32];  // bf16 A tile, row-major
  __shared__ __align__(16) unsigned short lB[256 * 32];      // bf16 B tile, TRANSPOSED lB[n][k]
  const int tid  = threadIdx.x;
  const int lane = tid & 31;
  const int wv   = tid >> 5;                  // 0..7
  const int half = lane >> 4;                 // 0/1
  const int l15  = lane & 15;
  const int m0   = blockIdx.x * (MT * 16);

  v8f acc0[MT], acc1[MT];
#pragma unroll
  for (int mt = 0; mt < MT; ++mt) {
    acc0[mt] = (v8f){0.f,0.f,0.f,0.f,0.f,0.f,0.f,0.f};
    acc1[mt] = (v8f){0.f,0.f,0.f,0.f,0.f,0.f,0.f,0.f};
  }

  for (int k0 = 0; k0 < K; k0 += 32) {
    // stage A tile (fused multi-source gather; zero-pad M/K remainders)
    for (int i = tid; i < MT * 16 * 32; i += 256) {
      int r = i >> 5, kk = i & 31;
      int gr = m0 + r, gk = k0 + kk;
      float v = 0.f;
      if (gr < M && gk < K) {
        if (gk < w0) {
          v = A0[(long)gr * w0 + gk];
        } else if (gk < w0 + 256) {
          int rr = i1 ? i1[gr] : gr;
          v = A1[(long)rr * 256 + (gk - w0)];
        } else {
          int rr = i2 ? i2[gr] : gr;
          v = A2[(long)rr * 256 + (gk - w0 - 256)];
        }
      }
      lA[i] = f2bf(v);
    }
    // stage B 32x256 transposed; coalesced over tid for each kk
#pragma unroll
    for (int kk = 0; kk < 32; ++kk) {
      int gk = k0 + kk;
      float v = (gk < K) ? W[(long)gk * 256 + tid] : 0.f;
      lB[tid * 32 + kk] = f2bf(v);
    }
    __syncthreads();

    // B fragments (this code's k order, same as A: lanes0-15 K=0..15, lanes16-31 K=16..31): 32 contiguous bytes
    Frag fb0, fb1;
    {
      const v4u* pb0 = (const v4u*)&lB[(wv * 32 + l15) * 32 + half * 16];
      fb0.q[0] = pb0[0]; fb0.q[1] = pb0[1];
      const v4u* pb1 = (const v4u*)&lB[(wv * 32 + 16 + l15) * 32 + half * 16];
      fb1.q[0] = pb1[0]; fb1.q[1] = pb1[1];
    }
#pragma unroll
    for (int mt = 0; mt < MT; ++mt) {
      // A fragment (16-bit A 16x32): two 16-byte chunks at kk = half*16 and half*16+8
      Frag fa;
      const unsigned short* base = &lA[(mt * 16 + l15) * 32];
      fa.q[0] = *(const v4u*)&base[half * 16];
      fa.q[1] = *(const v4u*)&base[half * 16 + 8];
      acc0[mt] = __builtin_amdgcn_wmma_f32_16x16x32_bf16(false, fa.v, false, fb0.v,
                                                         (short)0, acc0[mt], false, false);
      acc1[mt] = __builtin_amdgcn_wmma_f32_16x16x32_bf16(false, fa.v, false, fb1.v,
                                                         (short)0, acc1[mt], false, false);
    }
    __syncthreads();
  }

  // C/D layout: vgpr r -> row r + 8*half; col = lane&15
  const int c0 = wv * 32 + l15;
  const int c1 = c0 + 16;
#pragma unroll
  for (int mt = 0; mt < MT; ++mt) {
#pragma unroll
    for (int r = 0; r < 8; ++r) {
      int row = m0 + mt * 16 + r + 8 * half;
      if (row < M) {
        float v0 = acc0[mt][r], v1 = acc1[mt][r];
        if (bias)  { v0 += bias[c0]; v1 += bias[c1]; }
        if (addin) { v0 += addin[(long)row * 256 + c0]; v1 += addin[(long)row * 256 + c1]; }
        if (relu)  { v0 = fmaxf(v0, 0.f); v1 = fmaxf(v1, 0.f); }
        C[(long)row * 256 + c0] = v0;
        C[(long)row * 256 + c1] = v1;
      }
    }
  }
}

// ---------------- LayerNorm over 256 (optionally relu after) ----------------
__global__ __launch_bounds__(256)
void ln_kernel(const float* __restrict__ x, const float* __restrict__ gamma,
               const float* __restrict__ beta, float* __restrict__ y,
               int M, int postRelu) {
  __shared__ float red[256];
  int row = blockIdx.x;
  int d = threadIdx.x;
  float v = x[(long)row * 256 + d];
  red[d] = v; __syncthreads();
  for (int o = 128; o > 0; o >>= 1) { if (d < o) red[d] += red[d + o]; __syncthreads(); }
  float mean = red[0] * (1.f / 256.f);
  __syncthreads();
  float c = v - mean;
  red[d] = c * c; __syncthreads();
  for (int o = 128; o > 0; o >>= 1) { if (d < o) red[d] += red[d + o]; __syncthreads(); }
  float var = red[0] * (1.f / 256.f);
  float out = c * rsqrtf(var + EPS_LN) * gamma[d] + beta[d];
  if (postRelu) out = fmaxf(out, 0.f);
  y[(long)row * 256 + d] = out;
}

// step-0 edge features: dispBuf[e] = [dlat, dlon, ||featsS[s]-spat[p]||]  (stride 3)
__global__ void disp_kernel(const float* __restrict__ featsS, const float* __restrict__ spat,
                            const int* __restrict__ sphIdx, const int* __restrict__ spaIdx,
                            const float* __restrict__ dlat, const float* __restrict__ dlon,
                            float* __restrict__ dispBuf) {
  int e = blockIdx.x * blockDim.x + threadIdx.x;
  if (e >= N_EDGES) return;
  int s = sphIdx[e], p = spaIdx[e];
  float sum = 0.f;
  for (int d = 0; d < 256; ++d) {
    float diff = featsS[(long)s * 256 + d] - spat[(long)p * 256 + d];
    sum += diff * diff;
  }
  dispBuf[(long)e * 3 + 0] = dlat[e];
  dispBuf[(long)e * 3 + 1] = dlon[e];
  dispBuf[(long)e * 3 + 2] = sqrtf(sum);
}

// deterministic segment_sum: edges sorted by s in [offsets[s], offsets[s+1]); fill tail -> seg 0
__global__ __launch_bounds__(256)
void segsum_kernel(const float* __restrict__ edges, const int* __restrict__ offsets,
                   float* __restrict__ msgs) {
  int s = blockIdx.x, d = threadIdx.x;
  int total = offsets[N_SPHERE]; if (total > N_EDGES) total = N_EDGES;
  int b = offsets[s];     if (b > N_EDGES) b = N_EDGES;
  int e1 = offsets[s+1];  if (e1 > N_EDGES) e1 = N_EDGES;
  float sum = 0.f;
  for (int e = b; e < e1; ++e) sum += edges[(long)e * 256 + d];
  if (s == 0) for (int e = total; e < N_EDGES; ++e) sum += edges[(long)e * 256 + d];
  msgs[(long)s * 256 + d] = sum;
}

__global__ void nmean_kernel(const float* __restrict__ h, const int* __restrict__ nbr,
                             float* __restrict__ nm) {
  int t = blockIdx.x * blockDim.x + threadIdx.x;
  if (t >= N_SPHERE * 256) return;
  int s = t >> 8, d = t & 255;
  float sum = 0.f;
  for (int j = 0; j < 6; ++j) sum += h[(long)nbr[s*6+j] * 256 + d];
  nm[t] = sum * (1.f / 6.f);
}

// ---------------- host orchestration ----------------
extern "C" void kernel_launch(void* const* d_in, const int* in_sizes, int n_in,
                              void* d_out, int out_size, void* d_ws, size_t ws_size,
                              hipStream_t stream) {
  (void)in_sizes; (void)n_in; (void)out_size; (void)ws_size;
  const float* spatial_nodes = (const float*)d_in[0];
  const float* sphere_nodes  = (const float*)d_in[1];
  const float* W_proj = (const float*)d_in[2];
  const float* ew1a   = (const float*)d_in[3];
  const float* eb1a   = (const float*)d_in[4];
  const float* ew1b   = (const float*)d_in[5];
  const float* eb1b   = (const float*)d_in[6];
  const float* ew2    = (const float*)d_in[7];
  const float* eb2    = (const float*)d_in[8];
  const float* nw1    = (const float*)d_in[9];
  const float* nb1    = (const float*)d_in[10];
  const float* nw2    = (const float*)d_in[11];
  const float* nb2    = (const float*)d_in[12];
  const float* pws    = (const float*)d_in[13];
  const float* pbs    = (const float*)d_in[14];
  const float* pwn    = (const float*)d_in[15];
  const float* pbn    = (const float*)d_in[16];
  const float* elns   = (const float*)d_in[17];
  const float* nlns   = (const float*)d_in[18];
  const float* plns   = (const float*)d_in[19];
  const float* elno   = (const float*)d_in[20];
  const float* nlno   = (const float*)d_in[21];
  const float* plno   = (const float*)d_in[22];

  float* ws = (float*)d_ws;
  size_t off = 0;
  auto allocF = [&](size_t n) { float* p = ws + off; off += n; return p; };

  float* sg      = allocF(N_SPHERE * 6);
  float* pg      = allocF((size_t)N_SPATIAL * 6);
  float* pos     = allocF(N_SPHERE * 3);
  int*   counts  = (int*)allocF(N_SPHERE);
  int*   offs    = (int*)allocF(N_SPHERE + 1);
  int*   sphIdx  = (int*)allocF(N_EDGES);
  int*   spaIdx  = (int*)allocF(N_EDGES);
  int*   nbr     = (int*)allocF(N_SPHERE * 6);
  float* dlat    = allocF(N_EDGES);
  float* dlon    = allocF(N_EDGES);
  float* dispBuf = allocF((size_t)N_EDGES * 3);
  float* spat    = allocF((size_t)N_SPATIAL * LATENT);
  float* ehid    = allocF((size_t)N_EDGES * LATENT);
  float* edges   = allocF((size_t)N_EDGES * LATENT);
  float* feats   = allocF((size_t)N_SPHERE * LATENT);
  float* msgs    = allocF((size_t)N_SPHERE * LATENT);
  float* nhid    = allocF((size_t)N_SPHERE * LATENT);
  float* nm      = allocF((size_t)N_SPHERE * LATENT);
  float* t1      = allocF((size_t)N_SPHERE * LATENT);

  // ---- graph construction (deterministic, trig-free inner loops) ----
  sphere_geom_kernel<<<(N_SPHERE + 255) / 256, 256, 0, stream>>>(sg, pos);
  spatial_geom_kernel<<<(N_SPATIAL + 255) / 256, 256, 0, stream>>>(pg);
  count_edges_kernel<<<(N_SPHERE + 255) / 256, 256, 0, stream>>>(sg, pg, counts);
  scan_kernel<<<1, 1, 0, stream>>>(counts, offs);
  init_idx_kernel<<<(N_EDGES + 255) / 256, 256, 0, stream>>>(sphIdx, spaIdx);
  write_edges_kernel<<<(N_SPHERE + 255) / 256, 256, 0, stream>>>(sg, pg, offs, sphIdx, spaIdx);
  edge_geo_kernel<<<(N_EDGES + 255) / 256, 256, 0, stream>>>(sg, pg, sphIdx, spaIdx, dlat, dlon);
  neighbors_kernel<<<(N_SPHERE + 255) / 256, 256, 0, stream>>>(pos, nbr);

  // fused gather-GEMM launcher: MT=4 (64-row tiles) for big M, MT=1 for node-sized M
  auto gemmF = [&](const float* A0, int w0, const float* A1, const int* i1,
                   const float* A2, const int* i2, const float* W, const float* b,
                   const float* addin, float* C, int M, int K, int relu) {
    if (M >= 8192)
      gemm_fused<4><<<(M + 63) / 64, 256, 0, stream>>>(A0, w0, A1, i1, A2, i2, W, b,
                                                       addin, C, M, K, relu);
    else
      gemm_fused<1><<<(M + 15) / 16, 256, 0, stream>>>(A0, w0, A1, i1, A2, i2, W, b,
                                                       addin, C, M, K, relu);
  };
  auto gemm = [&](const float* A, const float* W, const float* b, const float* addin,
                  float* C, int M, int K, int relu) {
    gemmF(A, K, nullptr, nullptr, nullptr, nullptr, W, b, addin, C, M, K, relu);
  };
  auto ln = [&](const float* x, const float* g, const float* b, float* y, int M, int postRelu) {
    ln_kernel<<<M, 256, 0, stream>>>(x, g, b, y, M, postRelu);
  };

  hipMemcpyAsync(feats, sphere_nodes, (size_t)N_SPHERE * LATENT * sizeof(float),
                 hipMemcpyDeviceToDevice, stream);

  auto run_gnn = [&](int g, bool encoding) {
    gemm(spatial_nodes, W_proj + (size_t)g * N_FEAT * LATENT, nullptr, nullptr,
         spat, N_SPATIAL, N_FEAT, 0);
    for (int step = 0; step < 3; ++step) {
      int Kin = (step == 0) ? (3 + 2 * LATENT) : (3 * LATENT);
      const int*   sndIdx = encoding ? spaIdx : sphIdx;
      const float* sndSrc = encoding ? spat : feats;
      const int*   rcvIdx = encoding ? sphIdx : spaIdx;
      const float* rcvSrc = encoding ? feats : spat;
      const float* w1 = (step == 0) ? (ew1a + (size_t)g * 515 * LATENT)
                                    : (ew1b + (size_t)g * 768 * LATENT);
      const float* b1 = (step == 0) ? (eb1a + g * LATENT) : (eb1b + g * LATENT);
      if (step == 0) {
        disp_kernel<<<(N_EDGES + 255) / 256, 256, 0, stream>>>(feats, spat, sphIdx, spaIdx,
                                                               dlat, dlon, dispBuf);
        // e_in = [disp(3) | snd(256) | rcv(256)] gathered inside the GEMM
        gemmF(dispBuf, 3, sndSrc, sndIdx, rcvSrc, rcvIdx, w1, b1, nullptr,
              ehid, N_EDGES, Kin, 1);
      } else {
        // e_in = [edges(256) | snd(256) | rcv(256)] gathered inside the GEMM
        gemmF(edges, 256, sndSrc, sndIdx, rcvSrc, rcvIdx, w1, b1, nullptr,
              ehid, N_EDGES, Kin, 1);
      }
      ln(ehid, elns + g * LATENT, elno + g * LATENT, ehid, N_EDGES, 0);
      gemm(ehid, ew2 + (size_t)g * LATENT * LATENT, eb2 + g * LATENT, nullptr,
           edges, N_EDGES, LATENT, 0);
      segsum_kernel<<<N_SPHERE, 256, 0, stream>>>(edges, offs, msgs);
      // node input = [feats(256) | msgs(256)] gathered inside the GEMM
      gemmF(feats, 256, msgs, nullptr, nullptr, nullptr,
            nw1 + (size_t)g * 2 * LATENT * LATENT, nb1 + g * LATENT, nullptr,
            nhid, N_SPHERE, 2 * LATENT, 1);
      ln(nhid, nlns + g * LATENT, nlno + g * LATENT, nhid, N_SPHERE, 0);
      gemm(nhid, nw2 + (size_t)g * LATENT * LATENT, nb2 + g * LATENT, nullptr,
           feats, N_SPHERE, LATENT, 0);
    }
  };

  // encoder
  run_gnn(0, true);

  // processor: h = relu(LN(h + h@pws + pbs + nm@pwn + pbn))
  for (int i = 0; i < 3; ++i) {
    nmean_kernel<<<(N_SPHERE * LATENT + 255) / 256, 256, 0, stream>>>(feats, nbr, nm);
    gemm(feats, pws + (size_t)i * LATENT * LATENT, pbs + i * LATENT, feats, t1,
         N_SPHERE, LATENT, 0);
    gemm(nm, pwn + (size_t)i * LATENT * LATENT, pbn + i * LATENT, t1, nhid,
         N_SPHERE, LATENT, 0);
    ln(nhid, plns + i * LATENT, plno + i * LATENT, feats, N_SPHERE, 1);
  }

  // decoder
  run_gnn(1, false);

  hipMemcpyAsync(d_out, feats, (size_t)N_SPHERE * LATENT * sizeof(float),
                 hipMemcpyDeviceToDevice, stream);
}
